// ContactForceModel_43525198578334
// MI455X (gfx1250) — matrix operand order repaired and checked by reading, not compiled, not run on hardware
//
#include <hip/hip_runtime.h>
#include <hip/hip_bf16.h>

typedef __attribute__((ext_vector_type(16))) _Float16 v16h;
typedef __attribute__((ext_vector_type(8)))  _Float16 v8h;
typedef __attribute__((ext_vector_type(8)))  float    v8f;
typedef __attribute__((ext_vector_type(4)))  float    v4f;
typedef __attribute__((ext_vector_type(4)))  unsigned uv4;
typedef __attribute__((ext_vector_type(4)))  int      iv4;
typedef __attribute__((ext_vector_type(8)))  int      iv8;

#define BATCH 512
#define TSEQ  128
#define DIN   64
#define HID   512
#define NG    2048      // 4*HID
#define NSPH  128

#if __has_builtin(__builtin_amdgcn_tensor_load_to_lds)
#define USE_TDM 1
#else
#define USE_TDM 0
#endif

#if __has_builtin(__builtin_amdgcn_sched_barrier)
#define SCHED_FENCE() __builtin_amdgcn_sched_barrier(0)
#else
#define SCHED_FENCE()
#endif

// ---------------------------------------------------------------------------
// TDM: load a 64-row x 32-col f16 tile (row stride 512 f16 in global) into LDS
// with padded rows of 40 f16 (pad 8 f16 = 4 DWORDs after every 16 DWORDs).
// ---------------------------------------------------------------------------
#if USE_TDM
__device__ __forceinline__ void tdm_load_tile_64x32(const _Float16* src, unsigned lds_addr) {
    unsigned long long ga = (unsigned long long)(size_t)src;
    uv4 g0;
    g0.x = 1u;                                             // count=1, user mode
    g0.y = lds_addr;                                       // lds_addr [63:32]
    g0.z = (unsigned)(ga & 0xFFFFFFFFu);                   // global_addr lo
    g0.w = (unsigned)((ga >> 32) & 0x01FFFFFFu) | (2u << 30); // addr[56:32], type=2
    iv8 g1;
    g1[0] = (1 << 16)     // data_size = 1 -> 2 bytes
          | (1 << 20)     // pad_enable
          | (3 << 22)     // pad_interval: code 3 = 16 DWORDs (32 f16)
          | (3 << 25);    // pad_amount:  code 3 = 4 DWORDs (8 f16)
    g1[1] = (int)(512u << 16);   // tensor_dim0 = 512
    g1[2] = (int)(512u << 16);   // tensor_dim1 = 512
    g1[3] = (32 << 16);          // tile_dim0 = 32
    g1[4] = 64;                  // tile_dim1 = 64
    g1[5] = 512;                 // tensor_dim0_stride = 512 elements
    g1[6] = 0;
    g1[7] = 0;
    iv4 gz = {0, 0, 0, 0};
#if defined(__clang_major__) && (__clang_major__ >= 23)
    iv8 gz8 = {0, 0, 0, 0, 0, 0, 0, 0};
    __builtin_amdgcn_tensor_load_to_lds(g0, g1, gz, gz, gz8, 0);
#else
    __builtin_amdgcn_tensor_load_to_lds(g0, g1, gz, gz, 0);
#endif
}
#endif

// ---------------------------------------------------------------------------
// Small helper kernels
// ---------------------------------------------------------------------------
__global__ void zero_state_kernel(float* __restrict__ c, _Float16* __restrict__ h) {
    int i = blockIdx.x * blockDim.x + threadIdx.x;
    c[i] = 0.0f;
    h[i] = (_Float16)0.0f;
}

__global__ void copy_carry_kernel(const float* __restrict__ cs, const _Float16* __restrict__ hs,
                                  float* __restrict__ cd, _Float16* __restrict__ hd) {
    int i = blockIdx.x * blockDim.x + threadIdx.x;
    cd[i] = cs[i];
    hd[i] = hs[i];
}

// Permute bias into gate-interleaved order: out[4j+g] = b[g*512+j]
__global__ void bias_perm_kernel(const float* __restrict__ b, float* __restrict__ out) {
    int n = blockIdx.x * blockDim.x + threadIdx.x;   // 0..2047
    out[n] = b[(n & 3) * HID + (n >> 2)];
}

// Re-pack a row-major (Ktot x N) f32 matrix (split across two sources at rowsA)
// into WMMA B-fragment order (f16). gate_perm: packed col n draws from source
// column (n&3)*512 + (n>>2) (gate-interleaved LSTM layout).
__global__ void pack_frag_kernel(const float* __restrict__ A, int rowsA,
                                 const float* __restrict__ Bsrc,
                                 int N, int Ktot, int gate_perm,
                                 _Float16* __restrict__ out) {
    int idx = blockIdx.x * blockDim.x + threadIdx.x;
    if (idx >= Ktot * N) return;
    int i   = idx & 15;
    int l   = (idx >> 4) & 31;
    int rem = idx >> 9;
    int ntiles = N >> 4;
    int nt = rem % ntiles;
    int kb = rem / ntiles;
    int k  = (kb << 5) + ((i >> 3) << 4) + ((l >> 4) << 3) + (i & 7);
    int n  = (nt << 4) + (l & 15);
    if (gate_perm) n = (n & 3) * HID + (n >> 2);
    float v = (k < rowsA) ? A[(size_t)k * N + n] : Bsrc[(size_t)(k - rowsA) * N + n];
    out[idx] = (_Float16)v;
}

__device__ __forceinline__ float sigmoid_(float x) {
    return 1.0f / (1.0f + __expf(-x));
}
__device__ __forceinline__ float tanh_(float x) {
    float e = __expf(-2.0f * x);
    return (1.0f - e) / (1.0f + e);
}

// ---------------------------------------------------------------------------
// Fused LSTM step: z = [x_t | h_in] @ Wcat + b (gate-interleaved columns),
// then gate math in-kernel -> c (f32, in/out), h_out (f16), optional ys_t.
//   block = 128 threads (4 waves), tile 64(M) x 128(N), K-blocked by 32.
//   A tile: double-buffered LDS filled by TDM (async, overlapped with WMMA).
//   B fragments: packed global loads (hot in L2).
// Each block owns j-range [cn0/4, cn0/4+32) for its 64 rows: no races.
// ---------------------------------------------------------------------------
__global__ __launch_bounds__(128)
void lstm_step_kernel(const float* __restrict__ x32, int xs32,
                      const _Float16* __restrict__ x16,
                      int Kx,
                      const _Float16* __restrict__ h_in,
                      const _Float16* __restrict__ packW,   // fragment-packed, gate-interleaved
                      const float* __restrict__ biasP,      // gate-interleaved bias
                      float* __restrict__ c,
                      _Float16* __restrict__ h_out,
                      _Float16* __restrict__ ys_t)
{
    __shared__ __align__(16) _Float16 A_lds[2][64][40];
    __shared__ __align__(16) float zs[64][128];

    const int tid  = threadIdx.x;
    const int lane = tid & 31;
    const int wv   = tid >> 5;
    const int m0   = wv * 16;
    const int lr   = lane & 15;
    const int hi   = lane >> 4;

    const int arow = tid >> 1;           // staging: 2 threads/row
    const int ak0  = (tid & 1) << 4;
    const int rowbase = blockIdx.x * 64;
    const int cn0  = blockIdx.y * 128;

    const int nkb  = (Kx + HID) >> 5;

    const _Float16* xrow = x16  + (size_t)rowbase * HID;
    const _Float16* hrow = h_in + (size_t)rowbase * HID;

#if USE_TDM
    const unsigned ldsb0 = (unsigned)(size_t)(&A_lds[0][0][0]);
    const unsigned ldsb1 = (unsigned)(size_t)(&A_lds[1][0][0]);
    // prologue: kick off chunk 0 if it is an f16 source
    if (wv == 0 && !(x32 && 0 < Kx)) {
        const _Float16* s0 = (0 < Kx) ? xrow : (hrow + (0 - Kx));
        tdm_load_tile_64x32(s0, ldsb0);
    }
#endif

    v8f acc[8];
    const v8f vzero = {0.f, 0.f, 0.f, 0.f, 0.f, 0.f, 0.f, 0.f};
#pragma unroll
    for (int nt = 0; nt < 8; ++nt) acc[nt] = vzero;

    for (int kb = 0; kb < nkb; ++kb) {
        const int kg = kb << 5;                       // uniform
        const int cur = kb & 1;
        const bool cur_manual = (x32 != nullptr) && (kg < Kx);

        if (cur_manual) {
            // layer-0 input: f32 -> f16 staging (vectorized)
            const float* s = x32 + (size_t)(rowbase + arow) * xs32 + kg + ak0;
#pragma unroll
            for (int j = 0; j < 4; ++j) {
                v4f t = *(const v4f*)(s + 4 * j);
#pragma unroll
                for (int q = 0; q < 4; ++q)
                    A_lds[cur][arow][ak0 + 4 * j + q] = (_Float16)t[q];
            }
        }

#if USE_TDM
        if (wv == 0) {
            const int kgn = (kb + 1) << 5;
            const bool next_tdm = ((kb + 1) < nkb) && !((x32 != nullptr) && (kgn < Kx));
            if (next_tdm) {
                const _Float16* sn = (kgn < Kx) ? (xrow + kgn) : (hrow + (kgn - Kx));
                tdm_load_tile_64x32(sn, ((kb + 1) & 1) ? ldsb1 : ldsb0);
            }
            if (!cur_manual) {
                if (next_tdm) __builtin_amdgcn_s_wait_tensorcnt((short)1);
                else          __builtin_amdgcn_s_wait_tensorcnt((short)0);
            }
        }
#else
        if (!cur_manual) {
            const _Float16* src = (kg < Kx) ? (xrow + kg) : (hrow + (kg - Kx));
            const v8h* s = (const v8h*)(src + (size_t)arow * HID + ak0);
            *(v8h*)&A_lds[cur][arow][ak0]     = s[0];
            *(v8h*)&A_lds[cur][arow][ak0 + 8] = s[1];
        }
#endif
        __syncthreads();

        // A fragment: two contiguous 16B LDS reads + register shuffle
        v8h alo = *(const v8h*)&A_lds[cur][m0 + lr][hi << 3];
        v8h ahi = *(const v8h*)&A_lds[cur][m0 + lr][16 + (hi << 3)];
        v16h af = __builtin_shufflevector(alo, ahi,
                    0, 1, 2, 3, 4, 5, 6, 7, 8, 9, 10, 11, 12, 13, 14, 15);

        // B fragments: group all 8 packed loads, then 8 back-to-back WMMAs
        const _Float16* wp = packW
            + (((size_t)kb * 128 + (size_t)(blockIdx.y * 8)) * 32 + lane) * 16;
        if ((kb + 1) < nkb)
            __builtin_prefetch(wp + (size_t)128 * 32 * 16, 0, 1);

        v16h bf[8];
#pragma unroll
        for (int nt = 0; nt < 8; ++nt)
            bf[nt] = *(const v16h*)(wp + nt * 512);
        SCHED_FENCE();
#pragma unroll
        for (int nt = 0; nt < 8; ++nt)
            acc[nt] = __builtin_amdgcn_wmma_f32_16x16x32_f16(
                false, af, false, bf[nt], (short)0, acc[nt], false, false);

        __syncthreads();   // all waves done with A_lds[cur] before TDM(kb+2) refills it
    }

    // ---- epilogue 1: pre-activations (+bias) into LDS ----
#pragma unroll
    for (int nt = 0; nt < 8; ++nt) {
        const int col = nt * 16 + lr;
        const float bv = biasP[cn0 + col];
#pragma unroll
        for (int r = 0; r < 8; ++r)
            zs[m0 + r + (hi << 3)][col] = acc[nt][r] + bv;
    }
    __syncthreads();

    // ---- epilogue 2: gate math; block owns 64 rows x 32 j ----
    const int jbase = cn0 >> 2;
#pragma unroll
    for (int it = 0; it < 16; ++it) {
        const int idx = it * 128 + tid;
        const int row = idx >> 5;
        const int jj  = idx & 31;
        v4f zq = *(const v4f*)&zs[row][jj << 2];   // (i,f,g,o) quad
        const size_t cidx = (size_t)(rowbase + row) * HID + jbase + jj;
        float cc = sigmoid_(zq[1]) * c[cidx] + sigmoid_(zq[0]) * tanh_(zq[2]);
        float hh = sigmoid_(zq[3]) * tanh_(cc);
        c[cidx] = cc;
        _Float16 hf = (_Float16)hh;
        h_out[cidx] = hf;
        if (ys_t) ys_t[cidx] = hf;
    }
}

// ---------------------------------------------------------------------------
// Fused decoder MLP. All 64 rows of a block share one latent vector
// (b = row / S), so GEMM1's A-fragment loads straight from global.
// ---------------------------------------------------------------------------
__global__ __launch_bounds__(128)
void mlp_kernel(const float* __restrict__ spos,
                const float* __restrict__ srad,
                const _Float16* __restrict__ latent,   // (512,512) f16
                const _Float16* __restrict__ packM0,   // frag-packed (512,128)
                const float* __restrict__ W0,          // (516,128) for sphere rows 0..3
                const float* __restrict__ b0,
                const _Float16* __restrict__ packM1,   // frag-packed (128,64)
                const float* __restrict__ b1,
                const float* __restrict__ Wout,
                const float* __restrict__ bout,
                float* __restrict__ out)
{
    __shared__ __align__(16) _Float16 A2[64][136];
    __shared__ float z2s[64][72];

    const int tid  = threadIdx.x;
    const int lane = tid & 31;
    const int wv   = tid >> 5;
    const int m0   = wv * 16;
    const int lr   = lane & 15;
    const int hi   = lane >> 4;

    const _Float16* lat = latent + (size_t)(blockIdx.x >> 1) * HID;

    // ------------- GEMM1: K = 512 (latent part), N = 128 -------------
    v8f acc[8];
    const v8f vzero = {0.f, 0.f, 0.f, 0.f, 0.f, 0.f, 0.f, 0.f};
#pragma unroll
    for (int nt = 0; nt < 8; ++nt) acc[nt] = vzero;

#pragma unroll 4
    for (int kb = 0; kb < 16; ++kb) {
        v8h alo = *(const v8h*)(lat + (kb << 5) + (hi << 3));
        v8h ahi = *(const v8h*)(lat + (kb << 5) + 16 + (hi << 3));
        v16h af = __builtin_shufflevector(alo, ahi,
                    0, 1, 2, 3, 4, 5, 6, 7, 8, 9, 10, 11, 12, 13, 14, 15);
        const _Float16* wp = packM0 + (((size_t)kb * 8) * 32 + lane) * 16;
        v16h bf[8];
#pragma unroll
        for (int nt = 0; nt < 8; ++nt)
            bf[nt] = *(const v16h*)(wp + nt * 512);
        SCHED_FENCE();
#pragma unroll
        for (int nt = 0; nt < 8; ++nt)
            acc[nt] = __builtin_amdgcn_wmma_f32_16x16x32_f16(
                false, af, false, bf[nt], (short)0, acc[nt], false, false);
    }

    // epilogue: + b0 + rank-4 sphere contribution, relu -> A2 (f16)
#pragma unroll
    for (int r = 0; r < 8; ++r) {
        const int rowin = m0 + r + (hi << 3);
        const int rg = blockIdx.x * 64 + rowin;
        const int s  = rg & (NSPH - 1);
        const float p0 = spos[3 * s], p1 = spos[3 * s + 1], p2 = spos[3 * s + 2];
        const float p3 = srad[s];
#pragma unroll
        for (int nt = 0; nt < 8; ++nt) {
            const int col = nt * 16 + lr;
            float v = acc[nt][r] + b0[col]
                    + p0 * W0[col] + p1 * W0[128 + col]
                    + p2 * W0[256 + col] + p3 * W0[384 + col];
            A2[rowin][col] = (_Float16)(v > 0.f ? v : 0.f);
        }
    }
    __syncthreads();

    // ------------- GEMM2: K = 128, N = 64 -------------
    v8f acc2[4];
#pragma unroll
    for (int nt = 0; nt < 4; ++nt) acc2[nt] = vzero;

#pragma unroll
    for (int kb = 0; kb < 4; ++kb) {
        v8h alo = *(const v8h*)&A2[m0 + lr][(kb << 5) + (hi << 3)];
        v8h ahi = *(const v8h*)&A2[m0 + lr][(kb << 5) + 16 + (hi << 3)];
        v16h af = __builtin_shufflevector(alo, ahi,
                    0, 1, 2, 3, 4, 5, 6, 7, 8, 9, 10, 11, 12, 13, 14, 15);
        const _Float16* wp = packM1 + (((size_t)kb * 4) * 32 + lane) * 16;
        v16h bf[4];
#pragma unroll
        for (int nt = 0; nt < 4; ++nt)
            bf[nt] = *(const v16h*)(wp + nt * 512);
        SCHED_FENCE();
#pragma unroll
        for (int nt = 0; nt < 4; ++nt)
            acc2[nt] = __builtin_amdgcn_wmma_f32_16x16x32_f16(
                false, af, false, bf[nt], (short)0, acc2[nt], false, false);
    }

#pragma unroll
    for (int nt = 0; nt < 4; ++nt) {
        const int col = nt * 16 + lr;
        const float bv = b1[col];
#pragma unroll
        for (int r = 0; r < 8; ++r) {
            const int rowin = m0 + r + (hi << 3);
            float v = acc2[nt][r] + bv;
            z2s[rowin][col] = v > 0.f ? v : 0.f;
        }
    }
    __syncthreads();

    // ------------- final dot (64 -> 1) + softplus -------------
    if (tid < 64) {
        float accd = bout[0];
#pragma unroll
        for (int q = 0; q < 64; ++q) accd += z2s[tid][q] * Wout[q];
        float sp = fmaxf(accd, 0.f) + log1pf(__expf(-fabsf(accd)));
        out[(size_t)blockIdx.x * 64 + tid] = sp;
    }
}

// ---------------------------------------------------------------------------
// Host-side launch
// ---------------------------------------------------------------------------
extern "C" void kernel_launch(void* const* d_in, const int* in_sizes, int n_in,
                              void* d_out, int out_size, void* d_ws, size_t ws_size,
                              hipStream_t stream) {
    (void)in_sizes; (void)n_in; (void)out_size; (void)ws_size;

    const float* pose = (const float*)d_in[0];
    const float* spos = (const float*)d_in[1];
    const float* srad = (const float*)d_in[2];
    const float* Wx0  = (const float*)d_in[3];
    const float* Wh0  = (const float*)d_in[4];
    const float* bl0  = (const float*)d_in[5];
    const float* Wx1  = (const float*)d_in[6];
    const float* Wh1  = (const float*)d_in[7];
    const float* bl1  = (const float*)d_in[8];
    const float* W0   = (const float*)d_in[9];
    const float* b0   = (const float*)d_in[10];
    const float* W1   = (const float*)d_in[11];
    const float* b1   = (const float*)d_in[12];
    const float* Wout = (const float*)d_in[13];
    const float* bout = (const float*)d_in[14];
    float* out = (float*)d_out;

    // workspace layout (all regions 256B aligned by construction)
    char* p = (char*)d_ws;
    _Float16* packL0 = (_Float16*)p;  p += (size_t)576  * NG * 2;
    _Float16* packL1 = (_Float16*)p;  p += (size_t)1024 * NG * 2;
    _Float16* packM0 = (_Float16*)p;  p += (size_t)512 * 128 * 2;
    _Float16* packM1 = (_Float16*)p;  p += (size_t)128 * 64 * 2;
    float*    bP0  = (float*)p;       p += (size_t)NG * 4;
    float*    bP1  = (float*)p;       p += (size_t)NG * 4;
    _Float16* h0A  = (_Float16*)p;    p += (size_t)BATCH * HID * 2;
    _Float16* h0B  = (_Float16*)p;    p += (size_t)BATCH * HID * 2;
    _Float16* h1A  = (_Float16*)p;    p += (size_t)BATCH * HID * 2;
    _Float16* h1B  = (_Float16*)p;    p += (size_t)BATCH * HID * 2;
    float*    c0   = (float*)p;       p += (size_t)BATCH * HID * 4;
    float*    c1   = (float*)p;       p += (size_t)BATCH * HID * 4;
    _Float16* ys0  = (_Float16*)p;    // (T,B,H) f16, 64MB

    // --- pack weights (gate-interleaved for LSTM) + biases ---
    pack_frag_kernel<<<(576 * NG + 255) / 256, 256, 0, stream>>>(
        Wx0, 64, Wh0, NG, 576, 1, packL0);
    pack_frag_kernel<<<(1024 * NG + 255) / 256, 256, 0, stream>>>(
        Wx1, 512, Wh1, NG, 1024, 1, packL1);
    pack_frag_kernel<<<(512 * 128 + 255) / 256, 256, 0, stream>>>(
        W0 + 4 * 128, 512, W0 + 4 * 128, 128, 512, 0, packM0);
    pack_frag_kernel<<<(128 * 64 + 255) / 256, 256, 0, stream>>>(
        W1, 128, W1, 64, 128, 0, packM1);
    bias_perm_kernel<<<NG / 256, 256, 0, stream>>>(bl0, bP0);
    bias_perm_kernel<<<NG / 256, 256, 0, stream>>>(bl1, bP1);

    // --- init carry ---
    zero_state_kernel<<<(BATCH * HID) / 256, 256, 0, stream>>>(c0, h0A);

    dim3 ggrid(BATCH / 64, NG / 128);   // 8 x 16

    // --- layer 0 over time (h ping-pong; T even -> final h in h0A) ---
    {
        const _Float16* hin = h0A;
        _Float16* hout = h0B;
        for (int t = 0; t < TSEQ; ++t) {
            lstm_step_kernel<<<ggrid, 128, 0, stream>>>(
                pose + (size_t)t * DIN, TSEQ * DIN, nullptr, DIN,
                hin, packL0, bP0, c0, hout, ys0 + (size_t)t * BATCH * HID);
            const _Float16* tmp = hin; hin = hout; hout = (_Float16*)tmp;
        }
    }

    // --- carry layer0 final state into layer1 (reference semantics) ---
    copy_carry_kernel<<<(BATCH * HID) / 256, 256, 0, stream>>>(c0, h0A, c1, h1A);

    // --- layer 1 over time ---
    {
        const _Float16* hin = h1A;
        _Float16* hout = h1B;
        for (int t = 0; t < TSEQ; ++t) {
            lstm_step_kernel<<<ggrid, 128, 0, stream>>>(
                nullptr, 0, ys0 + (size_t)t * BATCH * HID, HID,
                hin, packL1, bP1, c1, hout, nullptr);
            const _Float16* tmp = hin; hin = hout; hout = (_Float16*)tmp;
        }
    }

    // --- fused decoder MLP; latent = final h1 (back in h1A after 128 swaps) ---
    mlp_kernel<<<(BATCH * NSPH) / 64, 128, 0, stream>>>(
        spos, srad, h1A, packM0, W0, b0, packM1, b1, Wout, bout, out);
}
